// BigBirdAttention_89936615178551
// MI455X (gfx1250) — matrix operand order repaired and checked by reading, compiled, not run on hardware
//
#include <hip/hip_runtime.h>
#include <hip/hip_bf16.h>

// ---------------------------------------------------------------------------
// BigBird block-sparse attention for MI455X (gfx1250, wave32, WMMA bf16).
// One 128-thread workgroup (4 waves) per (batch, head, q-block of 64 rows).
// Flash-attention streaming over the sparse key-block list.
// ---------------------------------------------------------------------------

typedef __attribute__((ext_vector_type(16))) __bf16 v16bf;
typedef __attribute__((ext_vector_type(8)))  float  v8f;

#define B_   2
#define H_   12
#define S_   4096
#define D_   64
#define NB_  64          // number of 64-wide blocks along S
#define NW_  62          // nb - 2 random-attention windows
#define R_   3           // random blocks per window
#define LDSS 72          // LDS row stride in bf16 elements (pad: 144B rows, 16B aligned)

__device__ __forceinline__ unsigned bfbits(float f) {
  // fp32 -> bf16 (round to nearest even), returned in low 16 bits
  unsigned u = __float_as_uint(f);
  return (u + 0x7FFFu + ((u >> 16) & 1u)) >> 16;
}
__device__ __forceinline__ unsigned packbf2(float a, float b) {
  return (bfbits(a) & 0xFFFFu) | (bfbits(b) << 16);
}

union FragU { v16bf v; uint4 u[2]; };

// 16 contiguous bf16 (two 16B LDS reads)
__device__ __forceinline__ v16bf ld16(const unsigned short* p) {
  FragU t; t.u[0] = *(const uint4*)p; t.u[1] = *(const uint4*)(p + 8); return t.v;
}
// two groups of 8 bf16 (A-matrix K-group layout)
__device__ __forceinline__ v16bf ld8x2(const unsigned short* p0, const unsigned short* p1) {
  FragU t; t.u[0] = *(const uint4*)p0; t.u[1] = *(const uint4*)p1; return t.v;
}
// 8 contiguous fp32 from global -> 8 packed bf16
__device__ __forceinline__ uint4 pack8f(const float* p) {
  float4 x = *(const float4*)p;
  float4 y = *(const float4*)(p + 4);
  uint4 u;
  u.x = packbf2(x.x, x.y); u.y = packbf2(x.z, x.w);
  u.z = packbf2(y.x, y.y); u.w = packbf2(y.z, y.w);
  return u;
}

__global__ __launch_bounds__(128) void bigbird_wmma_kernel(
    const float* __restrict__ q,   const float* __restrict__ k,
    const float* __restrict__ v,   const float* __restrict__ q_mask,
    const float* __restrict__ kv_mask, const float* __restrict__ band_mask,
    const float* __restrict__ q_block_mask, const float* __restrict__ kv_block_mask,
    const int*   __restrict__ random_attn, float* __restrict__ out)
{
  __shared__ unsigned short Ks[64 * LDSS];  // K block, [key][d], bf16
  __shared__ unsigned short Vt[64 * LDSS];  // V block transposed, [d][key], bf16
  __shared__ unsigned short Ps[64 * LDSS];  // P tile, [qrow][key], bf16

  const int qblk = blockIdx.x;
  const int h    = blockIdx.y;
  const int b    = blockIdx.z;
  const int tid  = threadIdx.x;
  const int wave = tid >> 5;
  const int lane = tid & 31;
  const int l16  = lane & 15;
  const int lh   = lane >> 4;

  const size_t bh = (size_t)(b * H_ + h) * S_;
  const float scale = 0.125f;  // 1/sqrt(64)

  // ---- Q fragments (A-matrix layout: M = lane%16, K groups split by half) ----
  const int arow = qblk * 64 + wave * 16 + l16;
  const float* qrow = q + (bh + arow) * D_;
  FragU qa0, qa1;
  qa0.u[0] = pack8f(qrow +      lh * 8);  qa0.u[1] = pack8f(qrow + 16 + lh * 8);
  qa1.u[0] = pack8f(qrow + 32 + lh * 8);  qa1.u[1] = pack8f(qrow + 48 + lh * 8);

  // ---- per-output-row constants (C/D layout: row = 8*half + component) ----
  float qm[8], qbm[8];
#pragma unroll
  for (int j = 0; j < 8; ++j) {
    int r = wave * 16 + lh * 8 + j;
    qm[j]  = q_mask[bh + (size_t)qblk * 64 + r];
    qbm[j] = q_block_mask[(size_t)(b * NB_ + qblk) * 64 + r];
  }

  float mrow[8], lrow[8];
  v8f acc[4];
  const v8f zero8 = {0.f, 0.f, 0.f, 0.f, 0.f, 0.f, 0.f, 0.f};
#pragma unroll
  for (int j = 0; j < 8; ++j) { mrow[j] = -INFINITY; lrow[j] = 0.f; }
#pragma unroll
  for (int t = 0; t < 4; ++t) acc[t] = zero8;

  const bool fullrow = (qblk == 0) || (qblk == NB_ - 1);
  const int nlist = fullrow ? NB_ : ((qblk == 1 || qblk == NB_ - 2) ? 7 : 8);

  for (int it = 0; it < nlist; ++it) {
    // ---- resolve key block + mask mode (0=kv_mask, 1=band_mask, 2=random) ----
    int kblk, mode = 0, bpos = 0;
    if (fullrow) {
      kblk = it;
    } else if (qblk == 1) {
      if (it < 3)       kblk = it;
      else if (it == 3) kblk = NB_ - 1;
      else { kblk = random_attn[(h * NW_ + 0) * R_ + (it - 4)]; mode = 2; }
    } else if (qblk == NB_ - 2) {
      if (it == 0)      kblk = 0;
      else if (it < 4)  kblk = NB_ - 4 + it;
      else { kblk = random_attn[(h * NW_ + (NW_ - 1)) * R_ + (it - 4)]; mode = 2; }
    } else {
      if (it == 0)      kblk = 0;
      else if (it < 4)  { kblk = qblk - 2 + it; mode = 1; bpos = it - 1; }
      else if (it == 4) kblk = NB_ - 1;
      else { kblk = random_attn[(h * NW_ + (qblk - 1)) * R_ + (it - 5)]; mode = 2; }
    }

    __syncthreads();  // protect Ks/Vt from previous iteration's readers

    // ---- stage K (row-major) and V^T into LDS as bf16 ----
    const float* kb = k + (bh + (size_t)kblk * 64) * D_;
    const float* vb = v + (bh + (size_t)kblk * 64) * D_;
#pragma unroll
    for (int i = 0; i < 8; ++i) {
      int idx = i * 128 + tid;
      int row = idx >> 4;
      int c4  = (idx & 15) * 4;
      float4 kx = *(const float4*)(kb + row * D_ + c4);
      unsigned* kd = (unsigned*)&Ks[row * LDSS + c4];
      kd[0] = packbf2(kx.x, kx.y);
      kd[1] = packbf2(kx.z, kx.w);
      float4 vx = *(const float4*)(vb + row * D_ + c4);
      Vt[(c4 + 0) * LDSS + row] = (unsigned short)bfbits(vx.x);
      Vt[(c4 + 1) * LDSS + row] = (unsigned short)bfbits(vx.y);
      Vt[(c4 + 2) * LDSS + row] = (unsigned short)bfbits(vx.z);
      Vt[(c4 + 3) * LDSS + row] = (unsigned short)bfbits(vx.w);
    }
    __syncthreads();

    // ---- S = Q K^T : 16x64 score tile per wave, 8 WMMA ----
    v8f sc[4];
#pragma unroll
    for (int nt = 0; nt < 4; ++nt) {
      const unsigned short* kr = &Ks[(nt * 16 + l16) * LDSS];  // B: N=lane%16
      v16bf b0 = ld8x2(kr +      lh * 8, kr + 16 + lh * 8);     // d 0..31
      v16bf b1 = ld8x2(kr + 32 + lh * 8, kr + 48 + lh * 8);     // d 32..63
      v8f c = zero8;
      c = __builtin_amdgcn_wmma_f32_16x16x32_bf16(false, qa0.v, false, b0, (short)0, c, false, false);
      c = __builtin_amdgcn_wmma_f32_16x16x32_bf16(false, qa1.v, false, b1, (short)0, c, false, false);
      sc[nt] = c;
    }

    // ---- scale + mask (-inf semantics, as in reference) ----
#pragma unroll
    for (int nt = 0; nt < 4; ++nt) {
      int ck = nt * 16 + l16;  // key index within block for this lane's column
      if (mode == 0) {
        float km = kv_mask[(size_t)b * S_ + (size_t)kblk * 64 + ck];
#pragma unroll
        for (int j = 0; j < 8; ++j)
          sc[nt][j] = (km != 0.f) ? sc[nt][j] * scale : -INFINITY;
      } else if (mode == 2) {
        float km = kv_block_mask[(size_t)(b * NB_ + kblk) * 64 + ck];
#pragma unroll
        for (int j = 0; j < 8; ++j)
          sc[nt][j] = (km * qbm[j] != 0.f) ? sc[nt][j] * scale : -INFINITY;
      } else {  // band
        const float* bmp = band_mask + ((size_t)b * (NB_ - 4) + (qblk - 2)) * 64 * 192;
#pragma unroll
        for (int j = 0; j < 8; ++j) {
          int r = wave * 16 + lh * 8 + j;
          float bm = bmp[(size_t)r * 192 + bpos * 64 + ck];
          sc[nt][j] = (bm != 0.f) ? sc[nt][j] * scale : -INFINITY;
        }
      }
    }

    // ---- online softmax (row reduce across 16-lane column groups) ----
#pragma unroll
    for (int j = 0; j < 8; ++j) {
      float t = fmaxf(fmaxf(sc[0][j], sc[1][j]), fmaxf(sc[2][j], sc[3][j]));
      t = fmaxf(t, __shfl_xor(t, 1, 32));
      t = fmaxf(t, __shfl_xor(t, 2, 32));
      t = fmaxf(t, __shfl_xor(t, 4, 32));
      t = fmaxf(t, __shfl_xor(t, 8, 32));
      float mnew  = fmaxf(mrow[j], t);
      float msub  = (mnew == -INFINITY) ? 0.f : mnew;
      float alpha = __expf(mrow[j] - msub);
      float psum  = 0.f;
#pragma unroll
      for (int nt = 0; nt < 4; ++nt) {
        float p = __expf(sc[nt][j] - msub);
        psum += p;
        // stash P (bf16) for the PV A-fragment re-layout
        Ps[(wave * 16 + lh * 8 + j) * LDSS + nt * 16 + l16] = (unsigned short)bfbits(p);
      }
      psum += __shfl_xor(psum, 1, 32);
      psum += __shfl_xor(psum, 2, 32);
      psum += __shfl_xor(psum, 4, 32);
      psum += __shfl_xor(psum, 8, 32);
      lrow[j] = lrow[j] * alpha + psum;
      mrow[j] = mnew;
#pragma unroll
      for (int nt = 0; nt < 4; ++nt) acc[nt][j] *= alpha;
    }

    // same-wave LDS write->read ordering before re-reading Ps in A-layout
    asm volatile("s_wait_dscnt 0" ::: "memory");

    // ---- O += P V : 8 WMMA per wave ----
    const unsigned short* pr = &Ps[(wave * 16 + l16) * LDSS];    // A: M=lane%16
    v16bf pa0 = ld8x2(pr +      lh * 8, pr + 16 + lh * 8);       // keys 0..31
    v16bf pa1 = ld8x2(pr + 32 + lh * 8, pr + 48 + lh * 8);       // keys 32..63
#pragma unroll
    for (int nt = 0; nt < 4; ++nt) {
      const unsigned short* vr = &Vt[(nt * 16 + l16) * LDSS];    // B: N=d=lane%16
      v16bf b0 = ld8x2(vr +      lh * 8, vr + 16 + lh * 8);       // keys 0..31
      v16bf b1 = ld8x2(vr + 32 + lh * 8, vr + 48 + lh * 8);       // keys 32..63
      acc[nt] = __builtin_amdgcn_wmma_f32_16x16x32_bf16(false, pa0, false, b0, (short)0, acc[nt], false, false);
      acc[nt] = __builtin_amdgcn_wmma_f32_16x16x32_bf16(false, pa1, false, b1, (short)0, acc[nt], false, false);
    }
  }

  // ---- finalize: O / l, apply q_mask, write fp32 output ----
#pragma unroll
  for (int j = 0; j < 8; ++j) {
    float rl = (lrow[j] > 0.f) ? (1.f / lrow[j]) : 0.f;
    int r = qblk * 64 + wave * 16 + lh * 8 + j;
    float* orow = out + (bh + r) * D_;
#pragma unroll
    for (int nt = 0; nt < 4; ++nt)
      orow[nt * 16 + l16] = acc[nt][j] * rl * qm[j];
  }
}

extern "C" void kernel_launch(void* const* d_in, const int* in_sizes, int n_in,
                              void* d_out, int out_size, void* d_ws, size_t ws_size,
                              hipStream_t stream) {
  (void)in_sizes; (void)n_in; (void)out_size; (void)d_ws; (void)ws_size;
  dim3 grid(NB_, H_, B_);   // (q-block, head, batch) = 64 x 12 x 2 workgroups
  bigbird_wmma_kernel<<<grid, 128, 0, stream>>>(
      (const float*)d_in[0],  // query
      (const float*)d_in[1],  // key
      (const float*)d_in[2],  // value
      (const float*)d_in[3],  // q_mask
      (const float*)d_in[4],  // kv_mask
      (const float*)d_in[5],  // band_mask
      (const float*)d_in[6],  // q_block_mask
      (const float*)d_in[7],  // kv_block_mask
      (const int*)d_in[8],    // random_attn
      (float*)d_out);
}
